// MultiHeadAttention_11922829213912
// MI455X (gfx1250) — matrix operand order repaired and checked by reading, not compiled, not run on hardware
//
#include <hip/hip_runtime.h>

// ---------------------------------------------------------------------------
// MI455X (gfx1250) multi-head attention, bf16 WMMA (v_wmma_f32_16x16x32_bf16)
// with f32 accumulation and f32 softmax. Memory-bound overall (537MB weights
// output): NT stores for streamed outputs, async global->LDS staging for
// bf16 tiles, L2 kept for the re-read q/k/v/concat intermediates.
// ---------------------------------------------------------------------------

typedef __attribute__((ext_vector_type(16))) __bf16 bf16x16;
typedef __attribute__((ext_vector_type(8)))  float  f32x8;

union Frag16 {
    bf16x16 v;
    uint4   q[2];
};

__device__ __forceinline__ unsigned short f2bf(float f) {
    union { float f; unsigned u; } x; x.f = f;
    unsigned r = (x.u + 0x7FFFu + ((x.u >> 16) & 1u)) >> 16;   // RNE
    return (unsigned short)r;
}
__device__ __forceinline__ unsigned pack2bf(float a, float b) {
    return (unsigned)f2bf(a) | ((unsigned)f2bf(b) << 16);
}

__device__ __forceinline__ f32x8 wmma_bf16(bf16x16 a, bf16x16 b, f32x8 c) {
    // (neg_a, A, neg_b, B, c_mod, C, reuse_a, reuse_b)
    return __builtin_amdgcn_wmma_f32_16x16x32_bf16(false, a, false, b,
                                                   (short)0, c, false, false);
}

// ---- CDNA5 async global->LDS copy (ASYNCcnt path), per-lane 16B ----
__device__ __forceinline__ void async_copy_b128(void* lds, const void* g) {
    unsigned loff = (unsigned)(unsigned long long)lds;   // low 32b = LDS offset
    asm volatile("global_load_async_to_lds_b128 %0, %1, off"
                 :: "v"(loff), "v"(g) : "memory");
}
__device__ __forceinline__ void wait_async() {
    asm volatile("s_wait_asynccnt 0x0" ::: "memory");
}

__device__ __forceinline__ float redmax16(float v) {
    v = fmaxf(v, __shfl_xor(v, 1, 32));
    v = fmaxf(v, __shfl_xor(v, 2, 32));
    v = fmaxf(v, __shfl_xor(v, 4, 32));
    v = fmaxf(v, __shfl_xor(v, 8, 32));
    return v;
}
__device__ __forceinline__ float redsum16(float v) {
    v += __shfl_xor(v, 1, 32);
    v += __shfl_xor(v, 2, 32);
    v += __shfl_xor(v, 4, 32);
    v += __shfl_xor(v, 8, 32);
    return v;
}

// ---------------------------------------------------------------------------
// GEMM: C = A(MxK) * B(KxN) + bias(N)
// OUT_MODE 0: bf16 scatter to (B,H,S,D)   (Q,K projections)
// OUT_MODE 1: f32 row-major (M,N), NT     (final projection)
// OUT_MODE 2: bf16 scatter to (B,H,D,S)   (V projection, pre-transposed)
// A_BF16   0: A is f32 (convert while staging); 1: A is bf16 (async staging)
// Assumes M%128==0, N%128==0, K%32==0, S=1024, H=16, D=64.
// ---------------------------------------------------------------------------
template <int OUT_MODE, int A_BF16>
__global__ __launch_bounds__(256)
void gemm_kernel(const void* __restrict__ Ap, const float* __restrict__ Bp,
                 const float* __restrict__ bias, void* __restrict__ Out,
                 int M, int N, int K) {
    __shared__ unsigned short lA[128][40];   // bf16 row-major, padded pitch
    __shared__ unsigned int   lB[128][20];   // [n][k/2] packed bf16 pairs

    const int tid  = threadIdx.x;
    const int lane = tid & 31;
    const int wave = tid >> 5;
    const int hf   = lane >> 4;   // half of the wave
    const int l16  = lane & 15;
    const int wm   = wave & 1;    // 2 row groups of 64
    const int wn   = wave >> 1;   // 4 col groups of 32

    const int m0 = blockIdx.y * 128;
    const int n0 = blockIdx.x * 128;

    f32x8 acc[4][2];
#pragma unroll
    for (int i = 0; i < 4; ++i)
#pragma unroll
        for (int j = 0; j < 2; ++j) acc[i][j] = (f32x8)0.0f;

    const int arow = tid >> 1;
    const int acg  = (tid & 1) * 16;
    const int kp   = tid >> 4;          // 0..15  (k pair index)
    const int nc   = (tid & 15) * 8;    // 0..120

    for (int kk = 0; kk < K; kk += 32) {
        // ---- stage A tile (128 x 32) as bf16 ----
        if (A_BF16) {
            const unsigned short* src =
                (const unsigned short*)Ap + (size_t)(m0 + arow) * K + kk + acg;
            async_copy_b128(&lA[arow][acg],     src);
            async_copy_b128(&lA[arow][acg + 8], src + 8);
        } else {
            const float* src = (const float*)Ap + (size_t)(m0 + arow) * K + kk + acg;
            float4 a0 = *(const float4*)(src);
            float4 a1 = *(const float4*)(src + 4);
            float4 a2 = *(const float4*)(src + 8);
            float4 a3 = *(const float4*)(src + 12);
            uint4 u0 = make_uint4(pack2bf(a0.x, a0.y), pack2bf(a0.z, a0.w),
                                  pack2bf(a1.x, a1.y), pack2bf(a1.z, a1.w));
            uint4 u1 = make_uint4(pack2bf(a2.x, a2.y), pack2bf(a2.z, a2.w),
                                  pack2bf(a3.x, a3.y), pack2bf(a3.z, a3.w));
            *(uint4*)&lA[arow][acg]     = u0;
            *(uint4*)&lA[arow][acg + 8] = u1;
        }
        // ---- stage B tile (32 x 128) transposed + k-pair packed ----
        {
            const float* b0 = Bp + (size_t)(kk + 2 * kp) * N + n0 + nc;
            const float* b1 = b0 + N;
            float4 x0 = *(const float4*)(b0);
            float4 x1 = *(const float4*)(b0 + 4);
            float4 y0 = *(const float4*)(b1);
            float4 y1 = *(const float4*)(b1 + 4);
            float rk [8] = {x0.x, x0.y, x0.z, x0.w, x1.x, x1.y, x1.z, x1.w};
            float rk1[8] = {y0.x, y0.y, y0.z, y0.w, y1.x, y1.y, y1.z, y1.w};
#pragma unroll
            for (int i = 0; i < 8; ++i)
                lB[nc + i][kp] = pack2bf(rk[i], rk1[i]);
        }
        if (kk + 32 < K) {  // pull next tiles toward the caches
            __builtin_prefetch((const char*)Ap +
                ((size_t)(m0 + arow) * K + kk + 32 + acg) * (A_BF16 ? 2 : 4), 0, 1);
            __builtin_prefetch(Bp + (size_t)(kk + 32 + 2 * kp) * N + n0 + nc, 0, 1);
        }
        if (A_BF16) wait_async();
        __syncthreads();

        // ---- fragments + 8 WMMAs per wave ----
        Frag16 af[4], bf[2];
#pragma unroll
        for (int i = 0; i < 4; ++i) {
            const int r = wm * 64 + i * 16 + l16;
            af[i].q[0] = *(const uint4*)&lA[r][8 * hf];
            af[i].q[1] = *(const uint4*)&lA[r][8 * hf + 16];
        }
#pragma unroll
        for (int j = 0; j < 2; ++j) {
            const int n = wn * 32 + j * 16 + l16;
            bf[j].q[0] = *(const uint4*)&lB[n][4 * hf];
            bf[j].q[1] = *(const uint4*)&lB[n][4 * hf + 8];
        }
#pragma unroll
        for (int i = 0; i < 4; ++i)
#pragma unroll
            for (int j = 0; j < 2; ++j)
                acc[i][j] = wmma_bf16(af[i].v, bf[j].v, acc[i][j]);
        __syncthreads();
    }

    // ---- epilogue: bias + scatter ----
#pragma unroll
    for (int j = 0; j < 2; ++j) {
        const int n  = n0 + wn * 32 + j * 16 + l16;
        const float bb = bias[n];
#pragma unroll
        for (int i = 0; i < 4; ++i) {
            const int mbase = m0 + wm * 64 + i * 16 + 8 * hf;
#pragma unroll
            for (int r = 0; r < 8; ++r) {
                const float v = acc[i][j][r] + bb;
                const int m = mbase + r;
                if (OUT_MODE == 1) {
                    __builtin_nontemporal_store(v, (float*)Out + (size_t)m * N + n);
                } else {
                    const int b_ = m >> 10, s = m & 1023;
                    const int h_ = n >> 6,  d = n & 63;
                    if (OUT_MODE == 0) {
                        ((unsigned short*)Out)[(((size_t)(b_ * 16 + h_)) * 1024 + s) * 64 + d] = f2bf(v);
                    } else {
                        ((unsigned short*)Out)[(((size_t)(b_ * 16 + h_)) * 64 + d) * 1024 + s] = f2bf(v);
                    }
                }
            }
        }
    }
}

// ---------------------------------------------------------------------------
// Attention: per (b,h) and 128-query-row tile.
// q,k: (B,H,S,64) bf16;  v: (B,H,64,S) bf16 (pre-transposed)
// weights: (B,H,S,S) f32 out (NT);  concat: (B,S,DM) bf16 out
// Two-pass flash softmax over 16 key tiles of 64.
// ---------------------------------------------------------------------------
__device__ __forceinline__ void attn_scores(
    const unsigned short (*lq)[72], const unsigned short (*lk)[72],
    int wm, int wn, int hf, int l16, f32x8 sc[4]) {
#pragma unroll
    for (int i = 0; i < 4; ++i) sc[i] = (f32x8)0.0f;
#pragma unroll
    for (int ks = 0; ks < 2; ++ks) {
        Frag16 bfr;
        bfr.q[0] = *(const uint4*)&lk[wn * 16 + l16][ks * 32 + 8 * hf];
        bfr.q[1] = *(const uint4*)&lk[wn * 16 + l16][ks * 32 + 8 * hf + 16];
#pragma unroll
        for (int i = 0; i < 4; ++i) {
            Frag16 afr;
            const int r = wm * 64 + i * 16 + l16;
            afr.q[0] = *(const uint4*)&lq[r][ks * 32 + 8 * hf];
            afr.q[1] = *(const uint4*)&lq[r][ks * 32 + 8 * hf + 16];
            sc[i] = wmma_bf16(afr.v, bfr.v, sc[i]);
        }
    }
#pragma unroll
    for (int i = 0; i < 4; ++i) sc[i] = sc[i] * 0.125f;   // 1/sqrt(64)
}

__global__ __launch_bounds__(256)
void attn_kernel(const unsigned short* __restrict__ qb,
                 const unsigned short* __restrict__ kb,
                 const unsigned short* __restrict__ vb,
                 float* __restrict__ weights,
                 unsigned short* __restrict__ concat) {
    __shared__ unsigned short lq[128][72];
    __shared__ unsigned short lk[64][72];
    __shared__ unsigned short lv[64][72];   // [d][key] (v pre-transposed)
    __shared__ unsigned short lp[128][72];
    __shared__ float rowm[128], rowl[128], rowinv[128];
    __shared__ float part[4][128];

    const int tid  = threadIdx.x;
    const int lane = tid & 31;
    const int wave = tid >> 5;
    const int hf   = lane >> 4;
    const int l16  = lane & 15;
    const int wm   = wave & 1;    // 2 q-row groups of 64
    const int wn   = wave >> 1;   // 4 key/d groups of 16

    const int bh = blockIdx.y;          // b*16 + h
    const int q0 = blockIdx.x * 128;

    // stage q tile (128 x 64) via async copies
    {
        const int row = tid >> 1, cg = (tid & 1) * 32;
        const unsigned short* src = qb + ((size_t)bh * 1024 + q0 + row) * 64 + cg;
        async_copy_b128(&lq[row][cg],      src);
        async_copy_b128(&lq[row][cg + 8],  src + 8);
        async_copy_b128(&lq[row][cg + 16], src + 16);
        async_copy_b128(&lq[row][cg + 24], src + 24);
    }
    if (tid < 128) { rowm[tid] = -3.0e38f; rowl[tid] = 0.0f; }
    wait_async();
    __syncthreads();

    const int krow = tid >> 2;
    const int kcg  = (tid & 3) * 16;

    // -------- pass 1: running row max / sum --------
    for (int kt = 0; kt < 16; ++kt) {
        const int k0 = kt * 64;
        {
            const unsigned short* src = kb + ((size_t)bh * 1024 + k0 + krow) * 64 + kcg;
            async_copy_b128(&lk[krow][kcg],     src);
            async_copy_b128(&lk[krow][kcg + 8], src + 8);
        }
        wait_async();
        __syncthreads();

        f32x8 sc[4];
        attn_scores(lq, lk, wm, wn, hf, l16, sc);

#pragma unroll
        for (int i = 0; i < 4; ++i)
#pragma unroll
            for (int r = 0; r < 8; ++r) {
                float v = redmax16(sc[i][r]);
                if (l16 == 0) part[wn][wm * 64 + i * 16 + 8 * hf + r] = v;
            }
        __syncthreads();
        if (tid < 128) {
            float tmax = fmaxf(fmaxf(part[0][tid], part[1][tid]),
                               fmaxf(part[2][tid], part[3][tid]));
            float mo = rowm[tid];
            float mn = fmaxf(mo, tmax);
            rowl[tid] *= __expf(mo - mn);
            rowm[tid] = mn;
        }
        __syncthreads();
#pragma unroll
        for (int i = 0; i < 4; ++i)
#pragma unroll
            for (int r = 0; r < 8; ++r) {
                const int row = wm * 64 + i * 16 + 8 * hf + r;
                float e = __expf(sc[i][r] - rowm[row]);
                e = redsum16(e);
                if (l16 == 0) part[wn][row] = e;
            }
        __syncthreads();
        if (tid < 128)
            rowl[tid] += part[0][tid] + part[1][tid] + part[2][tid] + part[3][tid];
        __syncthreads();
    }
    if (tid < 128) rowinv[tid] = 1.0f / rowl[tid];
    __syncthreads();

    // -------- pass 2: write weights (NT), accumulate att = P @ V --------
    f32x8 av[4];
#pragma unroll
    for (int i = 0; i < 4; ++i) av[i] = (f32x8)0.0f;

    for (int kt = 0; kt < 16; ++kt) {
        const int k0 = kt * 64;
        {
            const unsigned short* src = kb + ((size_t)bh * 1024 + k0 + krow) * 64 + kcg;
            async_copy_b128(&lk[krow][kcg],     src);
            async_copy_b128(&lk[krow][kcg + 8], src + 8);
            const unsigned short* vsrc = vb + ((size_t)bh * 64 + krow) * 1024 + k0 + kcg;
            async_copy_b128(&lv[krow][kcg],     vsrc);
            async_copy_b128(&lv[krow][kcg + 8], vsrc + 8);
        }
        wait_async();
        __syncthreads();

        f32x8 sc[4];
        attn_scores(lq, lk, wm, wn, hf, l16, sc);

#pragma unroll
        for (int i = 0; i < 4; ++i)
#pragma unroll
            for (int r = 0; r < 8; ++r) {
                const int row = wm * 64 + i * 16 + 8 * hf + r;
                const float p = __expf(sc[i][r] - rowm[row]) * rowinv[row];
                __builtin_nontemporal_store(
                    p, weights + ((size_t)bh * 1024 + q0 + row) * 1024 +
                           k0 + wn * 16 + l16);
                lp[row][wn * 16 + l16] = f2bf(p);
            }
        __syncthreads();

#pragma unroll
        for (int ks = 0; ks < 2; ++ks) {
            Frag16 bfr;
            bfr.q[0] = *(const uint4*)&lv[wn * 16 + l16][ks * 32 + 8 * hf];
            bfr.q[1] = *(const uint4*)&lv[wn * 16 + l16][ks * 32 + 8 * hf + 16];
#pragma unroll
            for (int i = 0; i < 4; ++i) {
                Frag16 afr;
                const int r = wm * 64 + i * 16 + l16;
                afr.q[0] = *(const uint4*)&lp[r][ks * 32 + 8 * hf];
                afr.q[1] = *(const uint4*)&lp[r][ks * 32 + 8 * hf + 16];
                av[i] = wmma_bf16(afr.v, bfr.v, av[i]);
            }
        }
        __syncthreads();
    }

    // merged-heads store: concat[b][s][h*64+d]
    const int b_ = bh >> 4, h_ = bh & 15;
#pragma unroll
    for (int i = 0; i < 4; ++i)
#pragma unroll
        for (int r = 0; r < 8; ++r) {
            const int s = q0 + wm * 64 + i * 16 + 8 * hf + r;
            concat[((size_t)b_ * 1024 + s) * 1024 + h_ * 64 + wn * 16 + l16] =
                f2bf(av[i][r]);
        }
}

// ---------------------------------------------------------------------------
extern "C" void kernel_launch(void* const* d_in, const int* in_sizes, int n_in,
                              void* d_out, int out_size, void* d_ws, size_t ws_size,
                              hipStream_t stream) {
    const float* Q  = (const float*)d_in[0];
    const float* K  = (const float*)d_in[1];
    const float* V  = (const float*)d_in[2];
    const float* Wq = (const float*)d_in[3];
    const float* bq = (const float*)d_in[4];
    const float* Wk = (const float*)d_in[5];
    const float* bk = (const float*)d_in[6];
    const float* Wv = (const float*)d_in[7];
    const float* bv = (const float*)d_in[8];
    const float* Wo = (const float*)d_in[9];
    const float* bo = (const float*)d_in[10];

    float* out     = (float*)d_out;                       // (B,S,DM)
    float* weights = out + (size_t)8 * 1024 * 1024;       // (B,H,S,S)

    unsigned short* qb = (unsigned short*)d_ws;           // (B,H,S,64) bf16
    unsigned short* kb = qb + (size_t)8 * 16 * 1024 * 64; // (B,H,S,64) bf16
    unsigned short* vb = kb + (size_t)8 * 16 * 1024 * 64; // (B,H,64,S) bf16
    unsigned short* cc = vb + (size_t)8 * 16 * 1024 * 64; // (B,S,DM)   bf16

    dim3 blk(256);
    dim3 g1(8, 64);   // N/128, M/128 for 8192x1024 GEMMs

    gemm_kernel<0, 0><<<g1, blk, 0, stream>>>(Q, Wq, bq, qb, 8192, 1024, 1024);
    gemm_kernel<0, 0><<<g1, blk, 0, stream>>>(K, Wk, bk, kb, 8192, 1024, 1024);
    gemm_kernel<2, 0><<<g1, blk, 0, stream>>>(V, Wv, bv, vb, 8192, 1024, 1024);

    attn_kernel<<<dim3(8, 128), blk, 0, stream>>>(qb, kb, vb, weights, cc);

    gemm_kernel<1, 1><<<g1, blk, 0, stream>>>(cc, Wo, bo, out, 8192, 1024, 1024);
}
